// SequentialConv_77558519432065
// MI455X (gfx1250) — matrix operand order repaired and checked by reading, not compiled, not run on hardware
//
#include <hip/hip_runtime.h>
#include <cmath>

typedef __attribute__((ext_vector_type(16))) __bf16 v16bf;
typedef __attribute__((ext_vector_type(8)))  float  v8f;

#define THREADS 512
#define NB 8
#define CH 64
#define HH 192
#define WW 256

constexpr int SLOT_COLS   = WW + 2;            // 258 columns (col 0 / 257 are zero pad)
constexpr int SLOT_HALVES = SLOT_COLS * CH;    // 16512 bf16 per row slot, layout [w][ci]
constexpr int WLDS_HALVES = 4 * 3 * 3 * 2 * 32 * 16; // 36864 bf16, WMMA-A swizzled weights
constexpr int SMEM_HALVES = WLDS_HALVES + 3 * SLOT_HALVES;
constexpr size_t SMEM_BYTES = (size_t)SMEM_HALVES * 2;   // 172800 B < 320 KB

__device__ __forceinline__ unsigned short f2bf(float f) {
  // round-to-nearest-even fp32 -> bf16 (inputs are finite normals here)
  unsigned u = __float_as_uint(f);
  u += 0x7FFFu + ((u >> 16) & 1u);
  return (unsigned short)(u >> 16);
}

__global__ __launch_bounds__(THREADS, 1)
void seqconv_down(const float* __restrict__ X, const float* __restrict__ Wt,
                  const float* __restrict__ bias, float* __restrict__ Out)
{
  extern __shared__ unsigned short smem[];
  unsigned short* sW   = smem;                 // [WLDS_HALVES]
  unsigned short* sRow = smem + WLDS_HALVES;   // [3][SLOT_HALVES]

  const int b    = blockIdx.x;        // one workgroup per batch: independent recurrences
  const int tid  = threadIdx.x;
  const int lane = tid & 31;
  const int wv   = tid >> 5;          // 16 waves
  const int hs   = lane >> 4;         // half-wave select (K-half / M-half per ISA layout)
  const int Nn   = lane & 15;         // N column within tile

  const float* Xb = X   + (size_t)b * CH * HH * WW;
  float*       Ob = Out + (size_t)b * CH * HH * WW;

  // ---- one-time: weights -> LDS in WMMA A-operand order, contiguous k = 16*half + j (matches B loads) ----
  // index e = (((ct*9 + kh*3 + kw)*2 + ks)*32 + lane)*16 + j
  for (int e = tid; e < WLDS_HALVES; e += THREADS) {
    int j  = e & 15;
    int ln = (e >> 4) & 31;
    int ks = (e >> 9) & 1;
    int r  = e >> 10;
    int kw = r % 3;
    int kh = (r / 3) % 3;
    int ct = r / 9;
    int M  = ln & 15;
    int K  = (ln >> 4) * 16 + j; // lanes0-15: K 0..15 ; lanes16-31: 16..31
    int co = ct * 16 + M;
    int ci = ks * 32 + K;
    sW[e] = f2bf(Wt[(co * CH + ci) * 9 + kh * 3 + kw]);
  }
  // zero the pad columns of all 3 ring slots (stay zero forever)
  for (int i = tid; i < 3 * 2 * CH; i += THREADS) {
    int s   = i / (2 * CH);
    int rem = i % (2 * CH);
    int col = (rem >> 6) ? (SLOT_COLS - 1) : 0;
    sRow[s * SLOT_HALVES + col * CH + (rem & 63)] = 0;
  }
  // ring init: slots 0..2 = original rows 0..2 (bf16, transposed [w][ci])
  for (int idx = tid; idx < 3 * CH * WW; idx += THREADS) {
    int s  = idx / (CH * WW);
    int r  = idx % (CH * WW);
    int ci = r & 63;
    int w  = r >> 6;
    sRow[s * SLOT_HALVES + (w + 1) * CH + ci] = f2bf(Xb[(ci * HH + s) * WW + w]);
  }
  // output rows 0..1 = passthrough originals
  for (int idx = tid; idx < 2 * CH * WW; idx += THREADS) {
    int w   = idx % WW;
    int t   = idx / WW;
    int row = t & 1;
    int ci  = t >> 1;
    Ob[(ci * HH + row) * WW + w] = Xb[(ci * HH + row) * WW + w];
  }
  __syncthreads();

  // wave tiling: 2 co-tiles x 2 n-tiles per wave -> 16 waves cover 4x16 output tiles
  const int coP = wv & 1;
  const int nP  = wv >> 1;

  for (int pos = 2; pos < HH; ++pos) {
    const int r2 = pos % 3;            // slot holding original row pos (kh=2 input)
    const int r1 = (pos + 2) % 3;      // updated row pos-1 (kh=1)
    const int r0 = (pos + 1) % 3;      // updated row pos-2 (kh=0); recycled for row pos+1
    const unsigned short* rowS[3] = { sRow + r0 * SLOT_HALVES,
                                      sRow + r1 * SLOT_HALVES,
                                      sRow + r2 * SLOT_HALVES };

    v8f acc00 = {}, acc01 = {}, acc10 = {}, acc11 = {};
    #pragma unroll
    for (int kh = 0; kh < 3; ++kh) {
      const unsigned short* rs = rowS[kh];
      #pragma unroll
      for (int kw = 0; kw < 3; ++kw) {
        #pragma unroll
        for (int ks = 0; ks < 2; ++ks) {
          // B operand: lane = column w, 16 consecutive ci -> one contiguous 32B LDS read
          int c0 = (nP * 2 + 0) * 16 + Nn + kw;   // +kw encodes the (kw-1) shift + pad offset
          int c1 = (nP * 2 + 1) * 16 + Nn + kw;
          v16bf B0 = *(const v16bf*)(rs + c0 * CH + ks * 32 + hs * 16);
          v16bf B1 = *(const v16bf*)(rs + c1 * CH + ks * 32 + hs * 16);
          int a0 = (((((coP * 2 + 0) * 9 + kh * 3 + kw) * 2 + ks) * 32) + lane) * 16;
          int a1 = (((((coP * 2 + 1) * 9 + kh * 3 + kw) * 2 + ks) * 32) + lane) * 16;
          v16bf A0 = *(const v16bf*)(sW + a0);
          v16bf A1 = *(const v16bf*)(sW + a1);
          acc00 = __builtin_amdgcn_wmma_f32_16x16x32_bf16(false, A0, false, B0, (short)0, acc00, false, false);
          acc01 = __builtin_amdgcn_wmma_f32_16x16x32_bf16(false, A0, false, B1, (short)0, acc01, false, false);
          acc10 = __builtin_amdgcn_wmma_f32_16x16x32_bf16(false, A1, false, B0, (short)0, acc10, false, false);
          acc11 = __builtin_amdgcn_wmma_f32_16x16x32_bf16(false, A1, false, B1, (short)0, acc11, false, false);
        }
      }
    }
    __syncthreads();   // all LDS reads of this row done before slots are overwritten

    // epilogue: y = acc + bias; out = X_orig(fp32) + tanh(y); feed bf16 back into ring
    v8f accs[2][2] = { {acc00, acc01}, {acc10, acc11} };
    #pragma unroll
    for (int a = 0; a < 2; ++a) {
      int ctile = coP * 2 + a;
      #pragma unroll
      for (int n = 0; n < 2; ++n) {
        int wcol = (nP * 2 + n) * 16 + Nn;
        #pragma unroll
        for (int v = 0; v < 8; ++v) {
          int co = ctile * 16 + hs * 8 + v;     // D-tile: VGPR v, half-wave hs -> M = hs*8+v
          float y = accs[a][n][v] + bias[co];
          float o = Xb[(co * HH + pos) * WW + wcol] + tanhf(y);
          Ob[(co * HH + pos) * WW + wcol] = o;
          sRow[r2 * SLOT_HALVES + (wcol + 1) * CH + co] = f2bf(o);
        }
      }
    }
    // stage next original row into the recycled slot r0 (== (pos+1)%3)
    if (pos + 1 < HH) {
      for (int idx = tid; idx < CH * WW; idx += THREADS) {
        int ci = idx & 63;
        int w  = idx >> 6;
        sRow[r0 * SLOT_HALVES + (w + 1) * CH + ci] =
            f2bf(Xb[(ci * HH + (pos + 1)) * WW + w]);
      }
      if (pos + 2 < HH) {   // hint L2/L0 for the row after next (global_prefetch_b8)
        __builtin_prefetch(Xb + ((size_t)(tid & 63) * HH + (pos + 2)) * WW, 0, 0);
      }
    }
    __syncthreads();
  }
}

extern "C" void kernel_launch(void* const* d_in, const int* in_sizes, int n_in,
                              void* d_out, int out_size, void* d_ws, size_t ws_size,
                              hipStream_t stream) {
  const float* X  = (const float*)d_in[0];   // (8,64,192,256) fp32
  const float* Wt = (const float*)d_in[1];   // (64,64,3,3)    fp32
  const float* bs = (const float*)d_in[2];   // (64,)          fp32
  float* Out = (float*)d_out;                // (8,64,192,256) fp32

  static_assert(SMEM_BYTES == 172800, "LDS budget");
  hipFuncSetAttribute((const void*)seqconv_down,
                      hipFuncAttributeMaxDynamicSharedMemorySize, (int)SMEM_BYTES);
  seqconv_down<<<dim3(NB), dim3(THREADS), SMEM_BYTES, stream>>>(X, Wt, bs, Out);
}
